// Transformer_85057532330564
// MI455X (gfx1250) — matrix operand order repaired and checked by reading, not compiled, not run on hardware
//
#include <hip/hip_runtime.h>

// ---------------------------------------------------------------------------
// Types for CDNA5 WMMA (wave32): v_wmma_f32_16x16x32_bf16
// ---------------------------------------------------------------------------
typedef __attribute__((ext_vector_type(16))) __bf16        bf16x16;
typedef __attribute__((ext_vector_type(8)))  float         f32x8;
typedef __attribute__((ext_vector_type(16))) unsigned short u16x16;

union U16BF { u16x16 u; bf16x16 b; uint4 q[2]; };
union PK8   { unsigned short s[8]; uint4 q; };

__device__ __forceinline__ unsigned short f2bf(float f) {
    unsigned int u = __float_as_uint(f);
    unsigned int r = u + 0x7FFFu + ((u >> 16) & 1u);   // round-to-nearest-even
    return (unsigned short)(r >> 16);
}
__device__ __forceinline__ float bf2f(unsigned short h) {
    return __uint_as_float(((unsigned int)h) << 16);
}

// A fragment: 16x32 (MxK) bf16, src row-major (K contiguous), rows at src + (lane%16)*ld.
// ISA layout: lanes 0-15 -> M=lane, K elems {k0+half*8 .. +7, k0+16+half*8 .. +7}.
__device__ __forceinline__ bf16x16 load_a_frag(const unsigned short* __restrict__ src,
                                               int ld, int k0) {
    int lane = threadIdx.x & 31;
    int m = lane & 15, half = lane >> 4;
    const unsigned short* p = src + (size_t)m * ld + k0 + half * 8;
    U16BF u;
    u.q[0] = *(const uint4*)p;          // K = k0 + half*8 .. +7
    u.q[1] = *(const uint4*)(p + 16);   // K = k0 + 16 + half*8 .. +7
    return u.b;
}

// B fragment: 32x16 (KxN) bf16 from a column-transposed buffer bt[n][k] (K contiguous).
// ISA layout: lanes 0-15 -> N=lane, K elems {k0+half*8 .. +7, k0+16+half*8 .. +7} (matches A).
__device__ __forceinline__ bf16x16 load_b_frag_t(const unsigned short* __restrict__ bt,
                                                 int ld, int k0) {
    int lane = threadIdx.x & 31;
    int n = lane & 15, half = lane >> 4;
    const unsigned short* p = bt + (size_t)n * ld + k0 + half * 8;
    U16BF u;
    u.q[0] = *(const uint4*)p;
    u.q[1] = *(const uint4*)(p + 16);
    return u.b;
}

__device__ __forceinline__ f32x8 wmma_bf16(bf16x16 a, bf16x16 b, f32x8 c) {
    return __builtin_amdgcn_wmma_f32_16x16x32_bf16(false, a, false, b,
                                                   (short)0, c, false, false);
}

// ---------------------------------------------------------------------------
// Problem dims
// ---------------------------------------------------------------------------
#define B_  64
#define C_  512
#define N_  1024
#define CQ_ 64
#define LEAKY 0.2f
#define EPS_ 1e-6f

// padded LDS row strides (ushort elements) to avoid bank conflicts
#define SS 1032   // score/beta rows [16][N_] + 8 pad
#define TS 520    // tT rows [16][C_] + 8 pad
#define XS 520    // xT rows [16][C_] + 8 pad (proj kernel staging)

// ---------------------------------------------------------------------------
// Kernel 0: fp32 -> bf16 conversion (x, Wq, Wk, Wv)
// ---------------------------------------------------------------------------
__global__ void convert_bf16_kernel(const float* __restrict__ src,
                                    unsigned short* __restrict__ dst, int n) {
    int i = blockIdx.x * blockDim.x + threadIdx.x;
    if (i < n) dst[i] = f2bf(src[i]);
}

// ---------------------------------------------------------------------------
// Kernel 1: q/k projections -> fT[b][n][64], gT[b][n][64] (bf16, q contiguous)
// The 512x16 fp32 x-tile is staged once into LDS, transposed + converted to
// bf16 (xT[n][c], c contiguous) so every B fragment afterwards is two
// 16-byte ds_load_b128s instead of 16 strided scalar global loads + cvt.
// 8 waves: waves 0-3 compute f (4 row tiles of 16), waves 4-7 compute g.
// ---------------------------------------------------------------------------
__global__ __launch_bounds__(256) void proj_kernel(
        const float* __restrict__ x,
        const unsigned short* __restrict__ Wq_bf,
        const unsigned short* __restrict__ Wk_bf,
        unsigned short* __restrict__ fT,
        unsigned short* __restrict__ gT) {
    __shared__ __align__(16) unsigned short xT[16 * XS];  // [n][c] bf16

    int b     = blockIdx.x >> 6;
    int nbase = (blockIdx.x & 63) * 16;
    int wave  = threadIdx.x >> 5;
    int lane  = threadIdx.x & 31;
    int l16 = lane & 15, half = lane >> 4;

    // ---- stage x tile: coalesced fp32 reads, transposed bf16 LDS writes ----
    const float* xb = x + ((size_t)b * C_) * N_ + nbase;
#pragma unroll
    for (int it = 0; it < (C_ * 16) / 256; ++it) {
        int idx = threadIdx.x + it * 256;
        int n = idx & 15, c = idx >> 4;
        xT[n * XS + c] = f2bf(xb[(size_t)c * N_ + n]);
    }
    __syncthreads();

    int qtile = wave & 3;
    const unsigned short* W = (wave < 4) ? Wq_bf : Wk_bf;
    unsigned short* dst     = (wave < 4) ? fT : gT;

    f32x8 acc = {};
#pragma unroll 4
    for (int k0 = 0; k0 < C_; k0 += 32) {
        bf16x16 a  = load_a_frag(W + (size_t)qtile * 16 * C_, C_, k0);
        bf16x16 bb = load_b_frag_t(xT, XS, k0);
        acc = wmma_bf16(a, bb, acc);
    }
    // D layout: lane -> (q = qtile*16 + r + 8*half, n = nbase + l16); store transposed.
    PK8 pk;
#pragma unroll
    for (int r = 0; r < 8; ++r) pk.s[r] = f2bf(acc[r]);
    size_t row = (size_t)b * N_ + nbase + l16;
    *(uint4*)(dst + row * CQ_ + qtile * 16 + 8 * half) = pk.q;
}

// ---------------------------------------------------------------------------
// Kernel 2: fused attention per (batch, 16-query tile)
//   scores = fT_keys x gT_queries (WMMA, K=64) -> leaky-relu -> LDS
//   column softmax over 1024 keys + mask renormalization (wave32 shfl)
//   t   = x_bf16 . beta   (WMMA, K=1024)  -> LDS (transposed)
//   out = Wv . t          (WMMA, K=512)   -> gamma*acc*mask + x
// ---------------------------------------------------------------------------
__global__ __launch_bounds__(256) void attn_kernel(
        const float* __restrict__ x,
        const float* __restrict__ mask,
        const unsigned short* __restrict__ xbf,
        const unsigned short* __restrict__ fT,
        const unsigned short* __restrict__ gT,
        const unsigned short* __restrict__ Wv_bf,
        const float* __restrict__ gamma,
        float* __restrict__ out) {
    __shared__ __align__(16) unsigned short s_bf[16 * SS]; // scores -> beta, [j][i]
    __shared__ __align__(16) unsigned short tT[16 * TS];   // t transposed, [j][e]

    int b     = blockIdx.x >> 6;
    int jbase = (blockIdx.x & 63) * 16;
    int wave  = threadIdx.x >> 5;
    int lane  = threadIdx.x & 31;
    int l16 = lane & 15, half = lane >> 4;

    const unsigned short* fTb = fT + (size_t)b * N_ * CQ_;
    const unsigned short* gTb = gT + ((size_t)b * N_ + jbase) * CQ_;
    const float* maskb        = mask + (size_t)b * N_;

    // ---- Phase 1: scores[key i, query j] = sum_q f[q,i] g[q,j], leaky-relu ----
    bf16x16 bq0 = load_b_frag_t(gTb, CQ_, 0);    // query fragment reused for all key tiles
    bf16x16 bq1 = load_b_frag_t(gTb, CQ_, 32);
    for (int kt = wave; kt < 64; kt += 8) {
        int keybase = kt * 16;
        f32x8 acc = {};
        acc = wmma_bf16(load_a_frag(fTb + (size_t)keybase * CQ_, CQ_, 0),  bq0, acc);
        acc = wmma_bf16(load_a_frag(fTb + (size_t)keybase * CQ_, CQ_, 32), bq1, acc);
        PK8 pk;
#pragma unroll
        for (int r = 0; r < 8; ++r) {
            float s = acc[r];
            s = (s >= 0.0f) ? s : LEAKY * s;
            pk.s[r] = f2bf(s);
        }
        // D lane -> (i = keybase + r + 8*half, j = l16): store as s_bf[j][i]
        *(uint4*)(s_bf + (size_t)l16 * SS + keybase + 8 * half) = pk.q;
    }
    __syncthreads();

    // ---- Phase 2: per-query softmax over keys + mask renorm (2 rows per wave) ----
#pragma unroll
    for (int rr = 0; rr < 2; ++rr) {
        int j = wave * 2 + rr;
        unsigned short* row = s_bf + (size_t)j * SS;
        float mx = -3.4e38f;
        for (int i = lane; i < N_; i += 32) mx = fmaxf(mx, bf2f(row[i]));
#pragma unroll
        for (int o = 16; o > 0; o >>= 1) mx = fmaxf(mx, __shfl_xor(mx, o, 32));
        float Z = 0.0f, SM = 0.0f;
        for (int i = lane; i < N_; i += 32) {
            float p = __expf(bf2f(row[i]) - mx);
            Z  += p;
            SM += p * maskb[i];
            row[i] = f2bf(p);
        }
#pragma unroll
        for (int o = 16; o > 0; o >>= 1) {
            Z  += __shfl_xor(Z,  o, 32);
            SM += __shfl_xor(SM, o, 32);
        }
        // beta_i = p_i*mask_i / (SM + EPS*Z)  ==  (softmax*mask)/(sum+EPS)
        float inv = 1.0f / (SM + EPS_ * Z);
        for (int i = lane; i < N_; i += 32) {
            float p = bf2f(row[i]);
            row[i] = f2bf(p * maskb[i] * inv);
        }
    }
    __syncthreads();

    // ---- Phase 3: t[e,j] = sum_n x[e,n] * beta[n,j]  (4 e-tiles per wave) ----
    const unsigned short* xbb = xbf + (size_t)b * C_ * N_;
    for (int et = wave; et < 32; et += 8) {
        // prefetch next e-tile's rows into cache while this tile computes
        if (et + 8 < 32) {
            __builtin_prefetch(xbb + (size_t)(et + 8) * 16 * N_ + (size_t)l16 * N_, 0, 1);
        }
        f32x8 acc = {};
#pragma unroll 4
        for (int k0 = 0; k0 < N_; k0 += 32) {
            bf16x16 a  = load_a_frag(xbb + (size_t)et * 16 * N_, N_, k0);
            bf16x16 bb = load_b_frag_t(s_bf, SS, k0);
            acc = wmma_bf16(a, bb, acc);
        }
        PK8 pk;
#pragma unroll
        for (int r = 0; r < 8; ++r) pk.s[r] = f2bf(acc[r]);
        // D lane -> (e = et*16 + r + 8*half, j = l16): store as tT[j][e]
        *(uint4*)(tT + (size_t)l16 * TS + et * 16 + 8 * half) = pk.q;
    }
    __syncthreads();

    // ---- Phase 4: out[c,j] = gamma * (sum_e Wv[c,e] t[e,j]) * mask[j] + x[c,j] ----
    float gam = gamma[0];
    for (int ct = wave; ct < 32; ct += 8) {
        if (ct + 8 < 32) {
            __builtin_prefetch(Wv_bf + (size_t)(ct + 8) * 16 * C_ + (size_t)l16 * C_, 0, 1);
        }
        f32x8 acc = {};
#pragma unroll 4
        for (int k0 = 0; k0 < C_; k0 += 32) {
            bf16x16 a  = load_a_frag(Wv_bf + (size_t)ct * 16 * C_, C_, k0);
            bf16x16 bb = load_b_frag_t(tT, TS, k0);
            acc = wmma_bf16(a, bb, acc);
        }
        int m = jbase + l16;
        float mk = maskb[m];
#pragma unroll
        for (int r = 0; r < 8; ++r) {
            int c = ct * 16 + 8 * half + r;
            size_t idx = ((size_t)b * C_ + c) * N_ + m;
            out[idx] = gam * acc[r] * mk + x[idx];
        }
    }
}

// ---------------------------------------------------------------------------
// Launcher
// ---------------------------------------------------------------------------
extern "C" void kernel_launch(void* const* d_in, const int* in_sizes, int n_in,
                              void* d_out, int out_size, void* d_ws, size_t ws_size,
                              hipStream_t stream) {
    const float* x     = (const float*)d_in[0];  // [B, C, N]
    const float* mask  = (const float*)d_in[1];  // [B, N]
    const float* Wq    = (const float*)d_in[2];  // [Cq, C]
    const float* Wk    = (const float*)d_in[3];  // [Cq, C]
    const float* Wv    = (const float*)d_in[4];  // [C, C]
    const float* gamma = (const float*)d_in[5];  // [1]
    float* out = (float*)d_out;

    const size_t nX  = (size_t)B_ * C_ * N_;   // 33,554,432
    const size_t nQK = (size_t)CQ_ * C_;       // 32,768
    const size_t nV  = (size_t)C_ * C_;        // 262,144
    const size_t nFT = (size_t)B_ * N_ * CQ_;  // 4,194,304

    char* ws = (char*)d_ws;
    unsigned short* xbf   = (unsigned short*)ws;                 ws += nX  * 2;
    unsigned short* fT    = (unsigned short*)ws;                 ws += nFT * 2;
    unsigned short* gT    = (unsigned short*)ws;                 ws += nFT * 2;
    unsigned short* Wq_bf = (unsigned short*)ws;                 ws += nQK * 2;
    unsigned short* Wk_bf = (unsigned short*)ws;                 ws += nQK * 2;
    unsigned short* Wv_bf = (unsigned short*)ws;                 /* total ~84.5 MB */

    // Kernel 0: bf16 conversions
    convert_bf16_kernel<<<(unsigned)((nX  + 255) / 256), 256, 0, stream>>>(x,  xbf,   (int)nX);
    convert_bf16_kernel<<<(unsigned)((nQK + 255) / 256), 256, 0, stream>>>(Wq, Wq_bf, (int)nQK);
    convert_bf16_kernel<<<(unsigned)((nQK + 255) / 256), 256, 0, stream>>>(Wk, Wk_bf, (int)nQK);
    convert_bf16_kernel<<<(unsigned)((nV  + 255) / 256), 256, 0, stream>>>(Wv, Wv_bf, (int)nV);

    // Kernel 1: q/k projections (transposed bf16 layouts, LDS-staged x tile)
    proj_kernel<<<B_ * (N_ / 16), 256, 0, stream>>>(x, Wq_bf, Wk_bf, fT, gT);

    // Kernel 2: fused attention (scores -> softmax/mask -> t = x*beta -> Wv*t + epilogue)
    attn_kernel<<<B_ * (N_ / 16), 256, 0, stream>>>(x, mask, xbf, fT, gT, Wv_bf, gamma, out);
}
